// RFCNHead_81544249082330
// MI455X (gfx1250) — matrix operand order repaired and checked by reading, not compiled, not run on hardware
//
#include <hip/hip_runtime.h>

typedef _Float16 half8   __attribute__((ext_vector_type(8)));
typedef _Float16 half16  __attribute__((ext_vector_type(16)));
typedef float    f32x8   __attribute__((ext_vector_type(8)));

#define HH     50
#define WW     50
#define HWSZ   2500
#define NBATCH 2
#define C1     512
#define INC    1024
#define K2     4608   /* 512*9 */
#define C3     4096
#define MCLS   1029
#define MREG   4116

// -------------------- small prep kernels --------------------

// fused BN+bias affine: s=g/sqrt(v+eps), t=(bias-mean)*s + beta
__global__ void k_affine(const float* __restrict__ g, const float* __restrict__ beta,
                         const float* __restrict__ mean, const float* __restrict__ var,
                         const float* __restrict__ bias,
                         float* __restrict__ s, float* __restrict__ t, int C)
{
    int c = blockIdx.x * blockDim.x + threadIdx.x;
    if (c >= C) return;
    float sv = g[c] * rsqrtf(var[c] + 1e-5f);
    s[c] = sv;
    t[c] = (bias[c] - mean[c]) * sv + beta[c];
}

// elementwise f32 -> f16
__global__ void k_cvt_f16(const float* __restrict__ in, _Float16* __restrict__ out, int n)
{
    int i = blockIdx.x * blockDim.x + threadIdx.x;
    if (i < n) out[i] = (_Float16)in[i];
}

// w2 [co][ci][3][3] f32 -> [co][q*512 + ci] f16  (q = kh*3+kw)
__global__ void k_w2_reorder(const float* __restrict__ w2, _Float16* __restrict__ out, int n)
{
    int i = blockIdx.x * blockDim.x + threadIdx.x;
    if (i >= n) return;
    int c  = i & 511;
    int r  = i >> 9;
    int q  = r % 9;
    int co = r / 9;
    out[co * K2 + q * C1 + c] = (_Float16)w2[(co * C1 + c) * 9 + q];
}

// feature [b][c][hw] f32 -> x0t [b][hw][c] f16
__global__ void k_feat_t(const float* __restrict__ f, _Float16* __restrict__ out, int n)
{
    int i = blockIdx.x * blockDim.x + threadIdx.x;
    if (i >= n) return;
    int c  = i & (INC - 1);
    int r  = i >> 10;
    int hw = r % HWSZ;
    int b  = r / HWSZ;
    out[i] = (_Float16)f[((size_t)(b * INC + c)) * HWSZ + hw];
}

// x1t [b][hw][c] f16 -> x2col [b][hw][q*512 + c] f16 with zero padding
__global__ void k_im2col(const _Float16* __restrict__ x1t, _Float16* __restrict__ out, int n)
{
    int i = blockIdx.x * blockDim.x + threadIdx.x;
    if (i >= n) return;
    int c  = i & 511;
    int r  = i >> 9;
    int q  = r % 9;
    r /= 9;
    int hw = r % HWSZ;
    int b  = r / HWSZ;
    int h = hw / WW, w = hw % WW;
    int hh = h + q / 3 - 1;
    int ww = w + q % 3 - 1;
    _Float16 v = (_Float16)0.0f;
    if (hh >= 0 && hh < HH && ww >= 0 && ww < WW)
        v = x1t[(((size_t)(b * HWSZ + hh * WW + ww)) << 9) + c];
    out[i] = v;   // i == (b*HW+hw)*4608 + q*512 + c
}

// -------------------- register-blocked WMMA GEMM --------------------
// C[M,N] = A[M,K] * Bt[N,K]^T.
// Wave computes (WM*16) x (WN*16); 8 waves arranged 2(M) x 4(N) per block.
// Epilogue: y = acc*scale[m] + shift[m], optional ReLU.
// Store: outT (f16, [N,M], next layer's B matrix) or outF (f32, [M,N]).
template<int WM>
__global__ void __launch_bounds__(256)
k_wmma_gemm(const _Float16* __restrict__ A, const _Float16* __restrict__ Bt,
            const float* __restrict__ scale, const float* __restrict__ shift,
            float* __restrict__ outF, _Float16* __restrict__ outT,
            int M, int N, int K, int relu)
{
    constexpr int WN = 2;
    constexpr int BM = 2 * WM * 16;   // block rows (64 or 128)
    constexpr int BN = 4 * WN * 16;   // 128 block cols
    const int tilesBM = (M + BM - 1) / BM;
    const int bm = blockIdx.x % tilesBM;
    const int bn = blockIdx.x / tilesBM;
    const int w  = threadIdx.x >> 5;        // 0..7, wave-uniform
    const int wm = w & 1;
    const int wn = w >> 1;
    const int lane = threadIdx.x & 31;
    const int lr = lane & 15;               // M index (A) / N index (B,C)
    const int lh = lane >> 4;               // K-half selector

    Bt += (size_t)blockIdx.y * N * K;

    const int row0 = bm * BM + wm * (WM * 16);
    const int col0 = bn * BN + wn * (WN * 16);

    // clamp OOB rows/cols to row 0: garbage only reaches D rows/cols that are
    // never stored, and EXEC stays all-ones for the WMMAs.
    const _Float16* ap[WM];
    const _Float16* bp[WN];
#pragma unroll
    for (int i = 0; i < WM; ++i) {
        int r = row0 + 16 * i + lr;
        ap[i] = A + (size_t)(r < M ? r : 0) * K + 16 * lh;
    }
#pragma unroll
    for (int j = 0; j < WN; ++j) {
        int c = col0 + 16 * j + lr;
        bp[j] = Bt + (size_t)(c < N ? c : 0) * K + 16 * lh;
    }

    f32x8 zero = {};
    f32x8 acc[WM][WN];
#pragma unroll
    for (int i = 0; i < WM; ++i)
#pragma unroll
        for (int j = 0; j < WN; ++j) acc[i][j] = zero;

#pragma unroll 2
    for (int kk = 0; kk < K; kk += 32) {
        half16 af[WM];
        half16 bf[WN];
#pragma unroll
        for (int i = 0; i < WM; ++i) {
            // A 16x32 f16 fragment: lane half lh holds contiguous K=kk+16lh..+15
            half8 a0 = *(const half8*)(ap[i] + kk);
            half8 a1 = *(const half8*)(ap[i] + kk + 8);
            af[i] = __builtin_shufflevector(a0, a1,
                        0,1,2,3,4,5,6,7,8,9,10,11,12,13,14,15);
        }
#pragma unroll
        for (int j = 0; j < WN; ++j) {
            // B 32x16 f16 fragment: lane half lh holds contiguous K=kk+16lh..+15
            bf[j] = *(const half16*)(bp[j] + kk);
        }
#pragma unroll
        for (int i = 0; i < WM; ++i)
#pragma unroll
            for (int j = 0; j < WN; ++j)
                acc[i][j] = __builtin_amdgcn_wmma_f32_16x16x32_f16(
                                false, af[i], false, bf[j], (short)0,
                                acc[i][j], false, false);
    }

    if (outF) outF += (size_t)blockIdx.y * M * N;
    if (outT) outT += (size_t)blockIdx.y * N * M;
#pragma unroll
    for (int i = 0; i < WM; ++i) {
#pragma unroll
        for (int jj = 0; jj < 8; ++jj) {
            int row = row0 + 16 * i + 8 * lh + jj;
            if (row < M) {
                float s = scale ? scale[row] : 1.0f;
                float t = shift ? shift[row] : 0.0f;
#pragma unroll
                for (int j = 0; j < WN; ++j) {
                    int col = col0 + 16 * j + lr;
                    if (col < N) {
                        float v = acc[i][j][jj] * s + t;
                        if (relu) v = fmaxf(v, 0.0f);
                        if (outT) outT[(size_t)col * M + row] = (_Float16)v;
                        else      outF[(size_t)row * N + col] = v;
                    }
                }
            }
        }
    }
}

// -------------------- PS-RoI pooling + bin mean --------------------
// map: [B][Cout*49][H][W] f32 ; rois: [R][5] ; out: [R][Cout]
__global__ void k_psroi(const float* __restrict__ map, const float* __restrict__ rois,
                        float* __restrict__ out, int Cout, int R)
{
    int idx = blockIdx.x * blockDim.x + threadIdx.x;
    if (idx >= R * Cout) return;
    int r  = idx / Cout;
    int co = idx % Cout;
    const float* roi = rois + r * 5;
    int b = (int)roi[0];
    const float sc = 1.0f / 16.0f;
    int rsw = (int)rintf(roi[1] * sc);   // round-half-even, matches jnp.round
    int rsh = (int)rintf(roi[2] * sc);
    int rew = (int)rintf(roi[3] * sc);
    int reh = (int)rintf(roi[4] * sc);
    int rw = max(rew - rsw, 1);
    int rh = max(reh - rsh, 1);
    float bh = (float)rh / 7.0f;
    float bw = (float)rw / 7.0f;
    int Cin = Cout * 49;
    float accum = 0.0f;
    for (int ph = 0; ph < 7; ++ph) {
        int hs = min(max((int)floorf((float)ph * bh) + rsh, 0), HH);
        int he = min(max((int)ceilf((float)(ph + 1) * bh) + rsh, 0), HH);
        for (int pw = 0; pw < 7; ++pw) {
            int ws = min(max((int)floorf((float)pw * bw) + rsw, 0), WW);
            int we = min(max((int)ceilf((float)(pw + 1) * bw) + rsw, 0), WW);
            int c = co * 49 + ph * 7 + pw;
            const float* base = map + ((size_t)(b * Cin + c)) * HWSZ;
            float s = 0.0f;
            for (int h = hs; h < he; ++h)
                for (int w = ws; w < we; ++w)
                    s += base[h * WW + w];
            int area = (he - hs) * (we - ws);
            accum += (area > 0) ? s / (float)area : 0.0f;
        }
    }
    out[idx] = accum * (1.0f / 49.0f);
}

// -------------------- host driver --------------------

extern "C" void kernel_launch(void* const* d_in, const int* in_sizes, int n_in,
                              void* d_out, int out_size, void* d_ws, size_t ws_size,
                              hipStream_t stream)
{
    (void)in_sizes; (void)n_in; (void)out_size; (void)ws_size;

    const float* feature = (const float*)d_in[0];
    const float* propose = (const float*)d_in[1];
    const float* w1    = (const float*)d_in[2];
    const float* b1    = (const float*)d_in[3];
    const float* w2    = (const float*)d_in[4];
    const float* b2    = (const float*)d_in[5];
    const float* w3    = (const float*)d_in[6];
    const float* b3    = (const float*)d_in[7];
    const float* w_cls = (const float*)d_in[8];
    const float* b_cls = (const float*)d_in[9];
    const float* w_reg = (const float*)d_in[10];
    const float* b_reg = (const float*)d_in[11];
    const float* bn1_g = (const float*)d_in[12];
    const float* bn1_b = (const float*)d_in[13];
    const float* bn1_m = (const float*)d_in[14];
    const float* bn1_v = (const float*)d_in[15];
    const float* bn2_g = (const float*)d_in[16];
    const float* bn2_b = (const float*)d_in[17];
    const float* bn2_m = (const float*)d_in[18];
    const float* bn2_v = (const float*)d_in[19];
    const float* bn3_g = (const float*)d_in[20];
    const float* bn3_b = (const float*)d_in[21];
    const float* bn3_m = (const float*)d_in[22];
    const float* bn3_v = (const float*)d_in[23];

    char* ws = (char*)d_ws;
    size_t off = 0;
    auto take = [&](size_t bytes) -> char* {
        char* p = ws + off;
        off = (off + bytes + 255) & ~(size_t)255;
        return p;
    };

    float* s1 = (float*)take(C1 * 4);
    float* t1 = (float*)take(C1 * 4);
    float* s2 = (float*)take(C1 * 4);
    float* t2 = (float*)take(C1 * 4);
    float* s3 = (float*)take(C3 * 4);
    float* t3 = (float*)take(C3 * 4);
    _Float16* w1h   = (_Float16*)take((size_t)C1 * INC * 2);
    _Float16* w2h   = (_Float16*)take((size_t)C1 * K2 * 2);
    _Float16* w3h   = (_Float16*)take((size_t)C3 * C1 * 2);
    _Float16* wch   = (_Float16*)take((size_t)MCLS * C3 * 2);
    _Float16* wrh   = (_Float16*)take((size_t)MREG * C3 * 2);
    _Float16* x0t   = (_Float16*)take((size_t)NBATCH * HWSZ * INC * 2);
    _Float16* x1t   = (_Float16*)take((size_t)NBATCH * HWSZ * C1 * 2);
    _Float16* x2col = (_Float16*)take((size_t)NBATCH * HWSZ * K2 * 2);
    _Float16* x2t   = (_Float16*)take((size_t)NBATCH * HWSZ * C1 * 2);
    _Float16* x3t   = x2col;  // alias: x2col is dead once conv2 finishes
    float* clsmap = (float*)take((size_t)NBATCH * MCLS * HWSZ * 4);
    float* regmap = (float*)take((size_t)NBATCH * MREG * HWSZ * 4);

    const int TB = 256;
    auto gr = [&](int n) { return dim3((unsigned)((n + TB - 1) / TB)); };

    // BN/bias affine fusion params
    k_affine<<<gr(C1), TB, 0, stream>>>(bn1_g, bn1_b, bn1_m, bn1_v, b1, s1, t1, C1);
    k_affine<<<gr(C1), TB, 0, stream>>>(bn2_g, bn2_b, bn2_m, bn2_v, b2, s2, t2, C1);
    k_affine<<<gr(C3), TB, 0, stream>>>(bn3_g, bn3_b, bn3_m, bn3_v, b3, s3, t3, C3);

    // weights -> f16
    k_cvt_f16<<<gr(C1 * INC), TB, 0, stream>>>(w1, w1h, C1 * INC);
    k_w2_reorder<<<gr(C1 * K2), TB, 0, stream>>>(w2, w2h, C1 * K2);
    k_cvt_f16<<<gr(C3 * C1), TB, 0, stream>>>(w3, w3h, C3 * C1);
    k_cvt_f16<<<gr(MCLS * C3), TB, 0, stream>>>(w_cls, wch, MCLS * C3);
    k_cvt_f16<<<gr(MREG * C3), TB, 0, stream>>>(w_reg, wrh, MREG * C3);

    // feature -> [b][hw][c] f16
    k_feat_t<<<gr(NBATCH * HWSZ * INC), TB, 0, stream>>>(feature, x0t, NBATCH * HWSZ * INC);

    auto gemm = [&](const _Float16* Aw, const _Float16* Btm,
                    const float* sc, const float* sh,
                    float* oF, _Float16* oT, int M, int N, int K, int relu,
                    bool bigM) {
        int BM = bigM ? 128 : 64;
        int blocks = ((M + BM - 1) / BM) * ((N + 127) / 128);
        if (bigM)
            k_wmma_gemm<4><<<dim3((unsigned)blocks, NBATCH), 256, 0, stream>>>(
                Aw, Btm, sc, sh, oF, oT, M, N, K, relu);
        else
            k_wmma_gemm<2><<<dim3((unsigned)blocks, NBATCH), 256, 0, stream>>>(
                Aw, Btm, sc, sh, oF, oT, M, N, K, relu);
    };

    // conv1 1x1 + BN + ReLU  -> x1t (f16, transposed)
    gemm(w1h, x0t, s1, t1, nullptr, x1t, C1, HWSZ, INC, 1, false);
    // im2col for 3x3
    k_im2col<<<gr(NBATCH * HWSZ * 9 * C1), TB, 0, stream>>>(x1t, x2col, NBATCH * HWSZ * 9 * C1);
    // conv2 3x3 + BN + ReLU -> x2t
    gemm(w2h, x2col, s2, t2, nullptr, x2t, C1, HWSZ, K2, 1, false);
    // conv3 1x1 + BN + ReLU -> x3t
    gemm(w3h, x2t, s3, t3, nullptr, x3t, C3, HWSZ, C1, 1, true);
    // cls / reg heads (bias only) -> f32 maps in [b][c][h][w]
    gemm(wch, x3t, nullptr, b_cls, clsmap, nullptr, MCLS, HWSZ, C3, 0, true);
    gemm(wrh, x3t, nullptr, b_reg, regmap, nullptr, MREG, HWSZ, C3, 0, true);

    // PS-RoI pooling + 7x7 mean -> d_out = [scores 512x21 | deltas 512x84]
    float* out = (float*)d_out;
    k_psroi<<<gr(512 * 21), TB, 0, stream>>>(clsmap, propose, out, 21, 512);
    k_psroi<<<gr(512 * 84), TB, 0, stream>>>(regmap, propose, out + 512 * 21, 84, 512);
}
